// PerfectFeatureModel_48000554500592
// MI455X (gfx1250) — compile-verified, matrix operand order checked by reading, not run on hardware
//
#include <hip/hip_runtime.h>
#include <cmath>

typedef __attribute__((ext_vector_type(16))) __bf16 v16bf;
typedef __attribute__((ext_vector_type(8)))  __bf16 v8bf;
typedef __attribute__((ext_vector_type(8)))  float  v8f;

#define LRN 1024
#define HRN 2048

static inline unsigned cdivu(size_t a, size_t b) { return (unsigned)((a + b - 1) / b); }

// ---------------------------------------------------------------------------
// GEMM: C[M,Nc] = scale * (A[M,Kp] @ B[Nc,Kp]^T) + bias[n], optional ReLU.
// A, B bf16 row-major with contiguous K (lda/ldb elements), Kp multiple of 32.
// One wave per 32x32 tile: 2x2 register-blocked WMMA (4x v_wmma per K-step,
// 2x arithmetic intensity vs 16x16-per-wave, 4x fewer L2 panel re-reads).
// ---------------------------------------------------------------------------
__global__ __launch_bounds__(32)
void k_gemm_nt(const __bf16* __restrict__ A, int lda,
               const __bf16* __restrict__ B, int ldb,
               float* __restrict__ C, int ldc,
               int M, int Nc, int Kp,
               const float* __restrict__ bias, float scale, int doRelu)
{
    const int lane = threadIdx.x;
    const int half = lane >> 4;
    const int r    = lane & 15;
    const int m0 = blockIdx.y << 5;
    const int n0 = blockIdx.x << 5;

    int ma0 = m0 + r;      if (ma0 >= M)  ma0 = M - 1;
    int ma1 = m0 + 16 + r; if (ma1 >= M)  ma1 = M - 1;
    int nb0 = n0 + r;      if (nb0 >= Nc) nb0 = Nc - 1;
    int nb1 = n0 + 16 + r; if (nb1 >= Nc) nb1 = Nc - 1;

    const __bf16* arow0 = A + (size_t)ma0 * lda + half * 8;
    const __bf16* arow1 = A + (size_t)ma1 * lda + half * 8;
    const __bf16* brow0 = B + (size_t)nb0 * ldb + half * 8;
    const __bf16* brow1 = B + (size_t)nb1 * ldb + half * 8;

    v8f acc00 = {0.f,0.f,0.f,0.f,0.f,0.f,0.f,0.f};
    v8f acc01 = acc00, acc10 = acc00, acc11 = acc00;

    for (int k0 = 0; k0 < Kp; k0 += 32) {
        v8bf a0lo = *(const v8bf*)(arow0 + k0);
        v8bf a0hi = *(const v8bf*)(arow0 + k0 + 16);
        v8bf a1lo = *(const v8bf*)(arow1 + k0);
        v8bf a1hi = *(const v8bf*)(arow1 + k0 + 16);
        v8bf b0lo = *(const v8bf*)(brow0 + k0);
        v8bf b0hi = *(const v8bf*)(brow0 + k0 + 16);
        v8bf b1lo = *(const v8bf*)(brow1 + k0);
        v8bf b1hi = *(const v8bf*)(brow1 + k0 + 16);
        v16bf a0 = __builtin_shufflevector(a0lo, a0hi, 0,1,2,3,4,5,6,7,8,9,10,11,12,13,14,15);
        v16bf a1 = __builtin_shufflevector(a1lo, a1hi, 0,1,2,3,4,5,6,7,8,9,10,11,12,13,14,15);
        v16bf b0 = __builtin_shufflevector(b0lo, b0hi, 0,1,2,3,4,5,6,7,8,9,10,11,12,13,14,15);
        v16bf b1 = __builtin_shufflevector(b1lo, b1hi, 0,1,2,3,4,5,6,7,8,9,10,11,12,13,14,15);
        acc00 = __builtin_amdgcn_wmma_f32_16x16x32_bf16(false, a0, false, b0, (short)0, acc00, false, false);
        acc01 = __builtin_amdgcn_wmma_f32_16x16x32_bf16(false, a0, false, b1, (short)0, acc01, false, false);
        acc10 = __builtin_amdgcn_wmma_f32_16x16x32_bf16(false, a1, false, b0, (short)0, acc10, false, false);
        acc11 = __builtin_amdgcn_wmma_f32_16x16x32_bf16(false, a1, false, b1, (short)0, acc11, false, false);
    }

    const int cols[2] = {n0 + r, n0 + 16 + r};
    const v8f* accs[2][2] = {{&acc00, &acc01}, {&acc10, &acc11}};
#pragma unroll
    for (int bn = 0; bn < 2; ++bn) {
        const int col = cols[bn];
        if (col >= Nc) continue;
        const float badd = bias ? bias[col] : 0.f;
#pragma unroll
        for (int bm = 0; bm < 2; ++bm) {
            const v8f& a = *accs[bm][bn];
#pragma unroll
            for (int v = 0; v < 8; ++v) {
                const int row = m0 + bm * 16 + v + (half << 3);
                if (row < M) {
                    float val = a[v] * scale + badd;
                    if (doRelu) val = fmaxf(val, 0.f);
                    C[(size_t)row * ldc + col] = val;
                }
            }
        }
    }
}

// f32 [R,C] -> bf16 [R,Cpad], zero-padded columns
__global__ void k_convert_pad(const float* __restrict__ s, __bf16* __restrict__ d,
                              int R, int C, int Cpad)
{
    size_t idx = (size_t)blockIdx.x * blockDim.x + threadIdx.x;
    size_t tot = (size_t)R * Cpad;
    if (idx >= tot) return;
    int c = (int)(idx % Cpad);
    int r = (int)(idx / Cpad);
    d[idx] = (c < C) ? (__bf16)s[(size_t)r * C + c] : (__bf16)0.f;
}

// f32 [R,C] -> bf16 [C,Rpad] transposed, zero-padded rows
__global__ void k_transpose_convert(const float* __restrict__ s, __bf16* __restrict__ d,
                                    int R, int C, int Rpad)
{
    size_t idx = (size_t)blockIdx.x * blockDim.x + threadIdx.x;
    size_t tot = (size_t)C * Rpad;
    if (idx >= tot) return;
    int rr = (int)(idx % Rpad);
    size_t cc = idx / Rpad;
    d[idx] = (rr < R) ? (__bf16)s[(size_t)rr * C + cc] : (__bf16)0.f;
}

// f32 [Nn,F] head slice (coff..coff+C) -> bf16 [Nn,Cpad], zero-padded
__global__ void k_extract_head(const float* __restrict__ s, __bf16* __restrict__ d,
                               int Nn, int F, int coff, int C, int Cpad)
{
    size_t idx = (size_t)blockIdx.x * blockDim.x + threadIdx.x;
    size_t tot = (size_t)Nn * Cpad;
    if (idx >= tot) return;
    int c = (int)(idx % Cpad);
    int i = (int)(idx / Cpad);
    d[idx] = (c < C) ? (__bf16)s[(size_t)i * F + coff + c] : (__bf16)0.f;
}

// qe[h,i] = sum_c q[i, h*C+c] * We[h*C+c]
__global__ void k_qe(const float* __restrict__ q, const float* __restrict__ We,
                     float* __restrict__ qe, int Nn, int F, int H, int C)
{
    int idx = blockIdx.x * blockDim.x + threadIdx.x;
    if (idx >= H * Nn) return;
    int h = idx / Nn, i = idx % Nn;
    const float* qp = q + (size_t)i * F + h * C;
    const float* wp = We + (size_t)h * C;
    float sum = 0.f;
    for (int c = 0; c < C; ++c) sum += qp[c] * wp[c];
    qe[idx] = sum;
}

// Row softmax (block per row). scores already hold scale*(QK^T); edge term
// Ea[j,i]*qe[i]*scale added on the fly. Writes alpha (bf16) and s[i]=sum alpha*Ea.
__global__ __launch_bounds__(256)
void k_softmax_edge(const float* __restrict__ sc, const float* __restrict__ Ea, int ldea,
                    const float* __restrict__ qe, float scale,
                    __bf16* __restrict__ al, float* __restrict__ s_out,
                    int Nn, int hasEdge)
{
    __shared__ float red[256];
    const int i = blockIdx.x, t = threadIdx.x;
    const float qei = hasEdge ? qe[i] * scale : 0.f;
    const float* row = sc + (size_t)i * Nn;

    float m = -3.4e38f;
    for (int j = t; j < Nn; j += 256) {
        float v = row[j];
        if (hasEdge) v += Ea[(size_t)j * ldea + i] * qei;
        m = fmaxf(m, v);
    }
    red[t] = m; __syncthreads();
    for (int s = 128; s > 0; s >>= 1) { if (t < s) red[t] = fmaxf(red[t], red[t + s]); __syncthreads(); }
    m = red[0]; __syncthreads();

    float se = 0.f, sea = 0.f;
    for (int j = t; j < Nn; j += 256) {
        float ea = hasEdge ? Ea[(size_t)j * ldea + i] : 0.f;
        float e = __expf(row[j] + ea * qei - m);
        se += e; sea += e * ea;
    }
    red[t] = se; __syncthreads();
    for (int s = 128; s > 0; s >>= 1) { if (t < s) red[t] += red[t + s]; __syncthreads(); }
    se = red[0]; __syncthreads();
    red[t] = sea; __syncthreads();
    for (int s = 128; s > 0; s >>= 1) { if (t < s) red[t] += red[t + s]; __syncthreads(); }
    sea = red[0]; __syncthreads();

    const float inv = 1.f / se;
    for (int j = t; j < Nn; j += 256) {
        float ea = hasEdge ? Ea[(size_t)j * ldea + i] : 0.f;
        al[(size_t)i * Nn + j] = (__bf16)(__expf(row[j] + ea * qei - m) * inv);
    }
    if (hasEdge && t == 0) s_out[i] = sea * inv;
}

// out[i,f] += (hasEdge ? s[h(f),i]*We[f] : 0) + skip[i,f]
__global__ void k_edge_skip_add(float* __restrict__ tc, const float* __restrict__ sb,
                                const float* __restrict__ We, const float* __restrict__ sk,
                                int Nn, int F, int C, int hasEdge)
{
    size_t idx = (size_t)blockIdx.x * blockDim.x + threadIdx.x;
    size_t tot = (size_t)Nn * F;
    if (idx >= tot) return;
    int f = (int)(idx % F);
    int i = (int)(idx / F);
    float add = sk[idx];
    if (hasEdge) { int h = f / C; add += sb[(size_t)h * Nn + i] * We[f]; }
    tc[idx] += add;
}

// GraphNorm: per-feature over nodes (thread per feature column; coalesced rows)
__global__ void k_graphnorm(const float* __restrict__ x, float* __restrict__ y,
                            const float* __restrict__ w, const float* __restrict__ b,
                            const float* __restrict__ ms, int Nn, int F)
{
    int f = blockIdx.x * blockDim.x + threadIdx.x;
    if (f >= F) return;
    float sum = 0.f;
    for (int i = 0; i < Nn; ++i) sum += x[(size_t)i * F + f];
    float sub = (sum / Nn) * ms[f];
    float var = 0.f;
    for (int i = 0; i < Nn; ++i) { float o = x[(size_t)i * F + f] - sub; var += o * o; }
    var /= Nn;
    float inv = rsqrtf(var + 1e-5f) * w[f];
    float bb = b[f];
    for (int i = 0; i < Nn; ++i) y[(size_t)i * F + f] = (x[(size_t)i * F + f] - sub) * inv + bb;
}

// Row L2 normalize: f32 out + bf16 out
__global__ __launch_bounds__(256)
void k_l2rows(const float* __restrict__ x, float* __restrict__ y, __bf16* __restrict__ yb,
              int R, int C)
{
    __shared__ float red[256];
    int i = blockIdx.x, t = threadIdx.x;
    const float* row = x + (size_t)i * C;
    float ss = 0.f;
    for (int j = t; j < C; j += 256) { float v = row[j]; ss += v * v; }
    red[t] = ss; __syncthreads();
    for (int s = 128; s > 0; s >>= 1) { if (t < s) red[t] += red[t + s]; __syncthreads(); }
    float inv = rsqrtf(red[0]);
    for (int j = t; j < C; j += 256) {
        float v = row[j] * inv;
        y[(size_t)i * C + j] = v;
        yb[(size_t)i * C + j] = (__bf16)v;
    }
}

// L2-normalize columns of x[R,C] and emit transposed bf16 [C,R]
__global__ void k_l2cols_t(const float* __restrict__ x, __bf16* __restrict__ out, int R, int C)
{
    int f = blockIdx.x * blockDim.x + threadIdx.x;
    if (f >= C) return;
    float ss = 0.f;
    for (int i = 0; i < R; ++i) { float v = x[(size_t)i * C + f]; ss += v * v; }
    float inv = rsqrtf(ss);
    for (int i = 0; i < R; ++i) out[(size_t)f * R + i] = (__bf16)(x[(size_t)i * C + f] * inv);
}

// ---------------------------------------------------------------------------
static inline void launch_gemm(hipStream_t st, const __bf16* A, int lda,
                               const __bf16* B, int ldb, float* C, int ldc,
                               int M, int Nc, int Kp,
                               const float* bias, float scale, int relu)
{
    dim3 g((Nc + 31) / 32, (M + 31) / 32);
    k_gemm_nt<<<g, 32, 0, st>>>(A, lda, B, ldb, C, ldc, M, Nc, Kp, bias, scale, relu);
}

static void run_tconv(hipStream_t st, const __bf16* Ab, int Kpad, int Nn, int Kin,
                      int F, int H, int C, int Cpad,
                      const float* const* W /*Wq,bq,Wk,bk,Wv,bv,Ws,bs*/,
                      const float* We, const float* Ea, int ldea,
                      float* TC, float* Q, float* Kf, float* V, float* SKf, float* SC,
                      __bf16* WT, __bf16* VT, __bf16* QHB, __bf16* KHB, __bf16* AL,
                      float* QE, float* SB)
{
    const float iscale = 1.0f / sqrtf((float)C);
    float* outs[4] = {Q, Kf, V, SKf};
    for (int t = 0; t < 4; ++t) {
        k_transpose_convert<<<cdivu((size_t)F * Kpad, 256), 256, 0, st>>>(W[2 * t], WT, Kin, F, Kpad);
        launch_gemm(st, Ab, Kpad, WT, Kpad, outs[t], F, Nn, F, Kpad, W[2 * t + 1], 1.0f, 0);
    }
    const int hasEdge = (Ea != nullptr);
    if (hasEdge)
        k_qe<<<cdivu((size_t)H * Nn, 256), 256, 0, st>>>(Q, We, QE, Nn, F, H, C);
    k_transpose_convert<<<cdivu((size_t)F * Nn, 256), 256, 0, st>>>(V, VT, Nn, F, Nn);

    for (int h = 0; h < H; ++h) {
        k_extract_head<<<cdivu((size_t)Nn * Cpad, 256), 256, 0, st>>>(Q,  QHB, Nn, F, h * C, C, Cpad);
        k_extract_head<<<cdivu((size_t)Nn * Cpad, 256), 256, 0, st>>>(Kf, KHB, Nn, F, h * C, C, Cpad);
        launch_gemm(st, QHB, Cpad, KHB, Cpad, SC, Nn, Nn, Nn, Cpad, nullptr, iscale, 0);
        k_softmax_edge<<<Nn, 256, 0, st>>>(SC, Ea, ldea,
                                           hasEdge ? (QE + (size_t)h * Nn) : nullptr, iscale,
                                           AL, SB + (size_t)h * Nn, Nn, hasEdge);
        launch_gemm(st, AL, Nn, VT + (size_t)h * C * Nn, Nn, TC + h * C, F,
                    Nn, C, Nn, nullptr, 1.0f, 0);
    }
    k_edge_skip_add<<<cdivu((size_t)Nn * F, 256), 256, 0, st>>>(TC, SB, We, SKf, Nn, F, C, hasEdge);
}

extern "C" void kernel_launch(void* const* d_in, const int* in_sizes, int n_in,
                              void* d_out_v, int out_size, void* d_ws, size_t ws_size,
                              hipStream_t stream)
{
    (void)in_sizes; (void)n_in; (void)out_size; (void)ws_size;
    const float* X = (const float*)d_in[0];
    auto F32 = [&](int i) { return (const float*)d_in[i]; };

    char* base = (char*)d_ws;
    size_t off = 0;
    auto alloc = [&](size_t bytes) -> void* {
        void* p = base + off;
        off = (off + bytes + 255) & ~(size_t)255;
        return p;
    };
    const size_t MF = (size_t)4096 * 2048;        // max M*F over all layers
    float*  Q    = (float*) alloc(MF * 4);
    float*  Kf   = (float*) alloc(MF * 4);
    float*  V    = (float*) alloc(MF * 4);
    float*  SK   = (float*) alloc(MF * 4);
    float*  TC   = (float*) alloc(MF * 4);
    float*  SC   = (float*) alloc((size_t)4096 * 4096 * 4);
    __bf16* AL   = (__bf16*)alloc((size_t)4096 * 4096 * 2);
    __bf16* WT   = (__bf16*)alloc((size_t)4096 * 2048 * 2);
    __bf16* VT   = (__bf16*)alloc((size_t)4096 * 2048 * 2);
    __bf16* QHB  = (__bf16*)alloc((size_t)4096 * 512 * 2);
    __bf16* KHB  = (__bf16*)alloc((size_t)4096 * 512 * 2);
    __bf16* AB   = (__bf16*)alloc((size_t)4096 * 2048 * 2);
    float*  QE   = (float*) alloc((size_t)8 * 4096 * 4);
    float*  SB   = (float*) alloc((size_t)8 * 4096 * 4);
    float*  H1   = (float*) alloc((size_t)1024 * 2048 * 4);
    float*  H2   = (float*) alloc((size_t)1024 * 4096 * 4);
    float*  LRX  = (float*) alloc((size_t)1024 * 4096 * 4);
    __bf16* LRXB = (__bf16*)alloc((size_t)1024 * 4096 * 2);
    float*  U1   = (float*) alloc((size_t)4096 * 256 * 4);
    float*  U2   = (float*) alloc((size_t)4096 * 2048 * 4);
    __bf16* U2B  = (__bf16*)alloc((size_t)2048 * 4096 * 2);

    float* OUT    = (float*)d_out_v;
    float* HR_OUT = OUT;                                  // [2048,2048]
    float* LR_OUT = OUT + (size_t)HRN * HRN;              // [1024,1024]

    // ---- layer fl1: N=1024, Fin=1024, F=2048, H=4, C=512, edge=x ----
    k_convert_pad<<<cdivu((size_t)1024 * 1024, 256), 256, 0, stream>>>(X, AB, 1024, 1024, 1024);
    {
        const float* Wl[8] = {F32(1), F32(2), F32(3), F32(4), F32(5), F32(6), F32(7), F32(8)};
        run_tconv(stream, AB, 1024, 1024, 1024, 2048, 4, 512, 512, Wl, F32(9), X, 1024,
                  TC, Q, Kf, V, SK, SC, WT, VT, QHB, KHB, AL, QE, SB);
    }
    k_graphnorm<<<cdivu(2048, 256), 256, 0, stream>>>(TC, H1, F32(10), F32(11), F32(12), 1024, 2048);

    // ---- layer fl2: N=1024, Fin=2048, F=4096, H=8, C=512, edge=x ----
    k_convert_pad<<<cdivu((size_t)1024 * 2048, 256), 256, 0, stream>>>(H1, AB, 1024, 2048, 2048);
    {
        const float* Wl[8] = {F32(13), F32(14), F32(15), F32(16), F32(17), F32(18), F32(19), F32(20)};
        run_tconv(stream, AB, 2048, 1024, 2048, 4096, 8, 512, 512, Wl, F32(21), X, 1024,
                  TC, Q, Kf, V, SK, SC, WT, VT, QHB, KHB, AL, QE, SB);
    }
    k_graphnorm<<<cdivu(4096, 256), 256, 0, stream>>>(TC, H2, F32(22), F32(23), F32(24), 1024, 4096);

    // ---- lr_x = l2rows(h2); lr output = relu(lr_x @ lr_x^T) ----
    k_l2rows<<<1024, 256, 0, stream>>>(H2, LRX, LRXB, 1024, 4096);
    launch_gemm(stream, LRXB, 4096, LRXB, 4096, LR_OUT, 1024, 1024, 1024, 4096, nullptr, 1.0f, 1);

    // ---- layer up1: input lr_x^T [4096,1024], F=200, H=4, C=50 (Cpad=64) ----
    k_transpose_convert<<<cdivu((size_t)4096 * 1024, 256), 256, 0, stream>>>(LRX, AB, 1024, 4096, 1024);
    {
        const float* Wl[8] = {F32(25), F32(26), F32(27), F32(28), F32(29), F32(30), F32(31), F32(32)};
        run_tconv(stream, AB, 1024, 4096, 1024, 200, 4, 50, 64, Wl, nullptr, nullptr, 0,
                  TC, Q, Kf, V, SK, SC, WT, VT, QHB, KHB, AL, QE, SB);
    }
    k_graphnorm<<<cdivu(200, 256), 256, 0, stream>>>(TC, U1, F32(33), F32(34), F32(35), 4096, 200);

    // ---- layer up2: input u1 [4096,200] (Kpad=224), F=2048, H=4, C=512 ----
    k_convert_pad<<<cdivu((size_t)4096 * 224, 256), 256, 0, stream>>>(U1, AB, 4096, 200, 224);
    {
        const float* Wl[8] = {F32(36), F32(37), F32(38), F32(39), F32(40), F32(41), F32(42), F32(43)};
        run_tconv(stream, AB, 224, 4096, 200, 2048, 4, 512, 512, Wl, nullptr, nullptr, 0,
                  TC, Q, Kf, V, SK, SC, WT, VT, QHB, KHB, AL, QE, SB);
    }
    k_graphnorm<<<cdivu(2048, 256), 256, 0, stream>>>(TC, U2, F32(44), F32(45), F32(46), 4096, 2048);

    // ---- u2 = l2rows(gnorm(...).T): normalize columns of U2, transposed bf16 ----
    k_l2cols_t<<<cdivu(2048, 256), 256, 0, stream>>>(U2, U2B, 4096, 2048);

    // ---- hr = relu(u2 @ u2^T) ----
    launch_gemm(stream, U2B, 4096, U2B, 4096, HR_OUT, 2048, 2048, 2048, 4096, nullptr, 1.0f, 1);
}
